// MultiHeadedSelfAttention_48524540510668
// MI455X (gfx1250) — matrix operand order repaired and checked by reading, not compiled, not run on hardware
//
#include <hip/hip_runtime.h>

// Problem constants (fixed by the reference).
#define S_LEN 2048
#define D_DIM 1024
#define A_DIM 1024
#define H_NUM 16
#define HD    64

typedef __attribute__((ext_vector_type(16))) _Float16 v16h;
typedef __attribute__((ext_vector_type(8)))  _Float16 v8h;
typedef __attribute__((ext_vector_type(8)))  float    v8f;

__device__ __forceinline__ v8f wmma16(v16h a, v16h b, v8f c) {
  // D = A(16x32 f16) * B(32x16 f16) + C(16x16 f32)
  return __builtin_amdgcn_wmma_f32_16x16x32_f16(false, a, false, b, (short)0, c,
                                                false, false);
}

// Async DMA: 16 bytes global -> LDS, tracked on ASYNCcnt (no VGPR staging).
// GVS addressing: lds-offset VGPR, 32-bit global byte-offset VGPR, SGPR64 base.
__device__ __forceinline__ void async_g2l_b128(unsigned lds_off, unsigned g_off,
                                               const void* sbase) {
  asm volatile("global_load_async_to_lds_b128 %0, %1, %2"
               :: "v"(lds_off), "v"(g_off), "s"(sbase) : "memory");
}
__device__ __forceinline__ void wait_asynccnt0() {
  asm volatile("s_wait_asynccnt 0x0" ::: "memory");
}

// A fragment (16x32, f16). ISA layout: lanes 0-15 hold rows M=0..15 with
// K = 0..7 and 16..23 in elements; lanes 16-31 hold K = 8..15 and 24..31.
// Source is row-major [m][k] with `stride` halfs per row; all offsets 16B aligned.
__device__ __forceinline__ v16h frag_a(const _Float16* base, int stride, int lane) {
  const int m  = lane & 15;
  const int kb = (lane >> 4) << 3;
  const _Float16* p = base + m * stride + kb;
  v8h lo = *(const v8h*)(p);
  v8h hi = *(const v8h*)(p + 16);
  v16h r;
#pragma unroll
  for (int i = 0; i < 8; ++i) { r[i] = lo[i]; r[i + 8] = hi[i]; }
  return r;
}

// B fragment (32x16, f16) from [n][k]-major storage: lanes 0-15 hold
// columns N=0..15 with K=0..7 and 16..23 in elements; lanes 16-31 hold K=8..15 and 24..31.
__device__ __forceinline__ v16h frag_b(const _Float16* base, int stride, int lane) {
  const int n  = lane & 15;
  const int kb = (lane >> 4) << 3;
  const _Float16* p = base + n * stride + kb;
  v8h lo = *(const v8h*)(p);
  v8h hi = *(const v8h*)(p + 16);
  v16h r;
#pragma unroll
  for (int i = 0; i < 8; ++i) { r[i] = lo[i]; r[i + 8] = hi[i]; }
  return r;
}

#define LDA 40  // 32 + 8 halfs pad; keeps every fragment read 16B aligned

// ---------------------------------------------------------------------------
// Kernel 1: QKV = x @ w_in.T, scattered to f16 Q/K/V in [H][S][hd] layout.
// Block = 256 thr (8 waves as 4x2), tile 128(M) x 64(N), K-chunks of 32.
// The 64-wide N tile is exactly one head's q, k or v slice.
// fp32 sources need on-the-fly f16 conversion -> VGPR staging path.
// ---------------------------------------------------------------------------
__global__ __launch_bounds__(256)
void qkv_gemm_k(const float* __restrict__ x, const float* __restrict__ w_in,
                _Float16* __restrict__ qh, _Float16* __restrict__ kh,
                _Float16* __restrict__ vh) {
  __shared__ _Float16 lA[128 * LDA];
  __shared__ _Float16 lB[64 * LDA];
  const int tid = threadIdx.x, lane = tid & 31, wv = tid >> 5;
  const int wmi = wv & 3;        // 4 waves along M (32 rows each)
  const int wni = wv >> 2;       // 2 waves along N (32 cols each)
  const int gm0 = blockIdx.x * 128;
  const int gy  = blockIdx.y;    // 0..47 -> which/head
  const int n0  = gy * 64;
  v8f acc[2][2] = {};
  for (int kt = 0; kt < D_DIM; kt += 32) {
    __syncthreads();
    if (kt + 32 < D_DIM)  // pull next A chunk toward the caches
      __builtin_prefetch(x + (size_t)(gm0 + (tid >> 1)) * D_DIM + kt + 32, 0, 1);
#pragma unroll
    for (int j = 0; j < 4; ++j) {           // A: 128x32 fp32 -> f16 LDS
      const int f = tid + j * 256;
      const int r = f >> 3, c = (f & 7) << 2;
      const float4 d = *(const float4*)(x + (size_t)(gm0 + r) * D_DIM + kt + c);
      _Float16* p = &lA[r * LDA + c];
      p[0] = (_Float16)d.x; p[1] = (_Float16)d.y;
      p[2] = (_Float16)d.z; p[3] = (_Float16)d.w;
    }
#pragma unroll
    for (int j = 0; j < 2; ++j) {           // B: 64x32 fp32 -> f16 LDS ([n][k])
      const int f = tid + j * 256;
      const int r = f >> 3, c = (f & 7) << 2;
      const float4 d = *(const float4*)(w_in + (size_t)(n0 + r) * D_DIM + kt + c);
      _Float16* p = &lB[r * LDA + c];
      p[0] = (_Float16)d.x; p[1] = (_Float16)d.y;
      p[2] = (_Float16)d.z; p[3] = (_Float16)d.w;
    }
    __syncthreads();
    v16h af[2], bf[2];
#pragma unroll
    for (int mi = 0; mi < 2; ++mi)
      af[mi] = frag_a(&lA[(wmi * 32 + mi * 16) * LDA], LDA, lane);
#pragma unroll
    for (int ni = 0; ni < 2; ++ni)
      bf[ni] = frag_b(&lB[(wni * 32 + ni * 16) * LDA], LDA, lane);
#pragma unroll
    for (int mi = 0; mi < 2; ++mi)
#pragma unroll
      for (int ni = 0; ni < 2; ++ni)
        acc[mi][ni] = wmma16(af[mi], bf[ni], acc[mi][ni]);
  }
  // Scatter: which = q/k/v, h = head; fold hd^-0.5 = 0.125 into Q.
  const int which = gy >> 4, h = gy & 15;
  _Float16* dst = (which == 0) ? qh : ((which == 1) ? kh : vh);
  const float sc = (which == 0) ? 0.125f : 1.0f;
  const int col = lane & 15, hs = lane >> 4;
#pragma unroll
  for (int mi = 0; mi < 2; ++mi)
#pragma unroll
    for (int ni = 0; ni < 2; ++ni)
#pragma unroll
      for (int i = 0; i < 8; ++i) {
        const int row = gm0 + wmi * 32 + mi * 16 + i + 8 * hs;
        const int dd  = wni * 32 + ni * 16 + col;
        dst[((size_t)h * S_LEN + row) * HD + dd] = (_Float16)(acc[mi][ni][i] * sc);
      }
}

// ---------------------------------------------------------------------------
// Kernel 2: flash attention per head, base-2 online softmax.
// Block = 128 thr (4 waves), 64 query rows (16 per wave), key chunks of 64.
// K chunk copied by async DMA (ASYNCcnt); V transposed via VGPRs in parallel.
// ---------------------------------------------------------------------------
#define PAD 72
__global__ __launch_bounds__(128)
void attn_k(const _Float16* __restrict__ qh, const _Float16* __restrict__ kh,
            const _Float16* __restrict__ vh, _Float16* __restrict__ ctx) {
  __shared__ _Float16 lK[64 * PAD];        // [t][d] key chunk
  __shared__ _Float16 lVt[64 * PAD];       // [d][t] value chunk, transposed
  __shared__ _Float16 lP[4 * 16 * PAD];    // per-wave probs [row][t]
  const int tid = threadIdx.x, lane = tid & 31, wv = tid >> 5;
  const int h  = blockIdx.y;
  const int q0 = blockIdx.x * 64;
  const _Float16* Qb = qh + (size_t)h * S_LEN * HD;
  const _Float16* Kb = kh + (size_t)h * S_LEN * HD;
  const _Float16* Vb = vh + (size_t)h * S_LEN * HD;
  // Q fragments for this wave's 16 rows (hd=64 -> 2 K-chunks of 32), held in VGPRs.
  v16h qf[2];
#pragma unroll
  for (int kc = 0; kc < 2; ++kc)
    qf[kc] = frag_a(Qb + (size_t)(q0 + wv * 16) * HD + kc * 32, HD, lane);
  v8f acc[4] = {};
  float mrow[8], lrow[8];
#pragma unroll
  for (int i = 0; i < 8; ++i) { mrow[i] = -1e30f; lrow[i] = 0.f; }
  _Float16* Pw = &lP[wv * 16 * PAD];
  const int col = lane & 15, hs = lane >> 4;
  for (int t0 = 0; t0 < S_LEN; t0 += 64) {
    __syncthreads();  // previous iteration done with lK/lVt
    // K chunk: 64x64 f16, async global->LDS DMA (no VGPR round-trip).
#pragma unroll
    for (int j = 0; j < 4; ++j) {
      const int f = tid + j * 128;
      const int r = f >> 3, c = (f & 7) << 3;
      const unsigned loff = (unsigned)(uintptr_t)&lK[r * PAD + c];
      const unsigned goff = (unsigned)(((t0 + r) * HD + c) * sizeof(_Float16));
      async_g2l_b128(loff, goff, Kb);
    }
    // V chunk transposed into [d][t]; overlaps with the async K copy.
#pragma unroll
    for (int j = 0; j < 32; ++j) {
      const int e = tid + j * 128;
      const int r = e >> 6, d = e & 63;
      lVt[d * PAD + r] = Vb[(size_t)(t0 + r) * HD + d];
    }
    wait_asynccnt0();   // our K rows have landed in LDS
    __syncthreads();    // everyone's K/V visible
    // scores: 4 key-subtiles x 2 hd-chunks
    v8f s[4] = {};
#pragma unroll
    for (int nt = 0; nt < 4; ++nt)
#pragma unroll
      for (int kc = 0; kc < 2; ++kc)
        s[nt] = wmma16(qf[kc], frag_b(&lK[(nt * 16) * PAD + kc * 32], PAD, lane),
                       s[nt]);
    // Online base-2 softmax. C-fragment: element i -> row i+8*(lane/16),
    // the 16 columns live across the 16 lanes of each half -> xor-butterfly.
    float corr[8];
#pragma unroll
    for (int i = 0; i < 8; ++i) {
      float v = fmaxf(fmaxf(s[0][i], s[1][i]), fmaxf(s[2][i], s[3][i]));
      v = fmaxf(v, __shfl_xor(v, 1));
      v = fmaxf(v, __shfl_xor(v, 2));
      v = fmaxf(v, __shfl_xor(v, 4));
      v = fmaxf(v, __shfl_xor(v, 8));
      const float mn = fmaxf(mrow[i], v);
      corr[i] = exp2f(mrow[i] - mn);
      mrow[i] = mn;
      float rs = 0.f;
#pragma unroll
      for (int nt = 0; nt < 4; ++nt) {
        const float p = exp2f(s[nt][i] - mn);   // p = 2^(score - rowmax)
        s[nt][i] = p;
        rs += p;
      }
      rs += __shfl_xor(rs, 1);
      rs += __shfl_xor(rs, 2);
      rs += __shfl_xor(rs, 4);
      rs += __shfl_xor(rs, 8);
      lrow[i] = lrow[i] * corr[i] + rs;
    }
    // Rescale accumulator; round-trip P through LDS to reshape C-frag -> A-frag.
#pragma unroll
    for (int nt = 0; nt < 4; ++nt)
#pragma unroll
      for (int i = 0; i < 8; ++i) {
        acc[nt][i] *= corr[i];
        Pw[(i + 8 * hs) * PAD + nt * 16 + col] = (_Float16)s[nt][i];
      }
    __syncthreads();
    // ctx += P(16x64) @ V(64x64)
#pragma unroll
    for (int kc = 0; kc < 2; ++kc) {
      const v16h a = frag_a(&Pw[kc * 32], PAD, lane);
#pragma unroll
      for (int nt = 0; nt < 4; ++nt)
        acc[nt] = wmma16(a, frag_b(&lVt[(nt * 16) * PAD + kc * 32], PAD, lane),
                         acc[nt]);
    }
  }
  // Normalize and write ctx in [S][A] layout for the output GEMM.
#pragma unroll
  for (int i = 0; i < 8; ++i) {
    const float inv = 1.0f / lrow[i];
    const int row = q0 + wv * 16 + i + 8 * hs;
#pragma unroll
    for (int nt = 0; nt < 4; ++nt)
      ctx[(size_t)row * A_DIM + h * HD + nt * 16 + col] =
          (_Float16)(acc[nt][i] * inv);
  }
}

// ---------------------------------------------------------------------------
// Kernel 3: out = ctx @ w_out.T  (f16 A async-copied, fp32 B converted, f32 out)
// ---------------------------------------------------------------------------
__global__ __launch_bounds__(256)
void out_gemm_k(const _Float16* __restrict__ ctx, const float* __restrict__ w_out,
                float* __restrict__ out) {
  __shared__ _Float16 lA[128 * LDA];
  __shared__ _Float16 lB[64 * LDA];
  const int tid = threadIdx.x, lane = tid & 31, wv = tid >> 5;
  const int wmi = wv & 3, wni = wv >> 2;
  const int gm0 = blockIdx.x * 128;
  const int n0  = blockIdx.y * 64;
  v8f acc[2][2] = {};
  for (int kt = 0; kt < A_DIM; kt += 32) {
    __syncthreads();
    // A: 128x32 f16 pure copy -> async DMA to LDS (512 b128 transfers).
#pragma unroll
    for (int j = 0; j < 2; ++j) {
      const int f = tid + j * 256;
      const int r = f >> 2, c = (f & 3) << 3;
      const unsigned loff = (unsigned)(uintptr_t)&lA[r * LDA + c];
      const unsigned goff =
          (unsigned)((((size_t)(gm0 + r) * A_DIM) + kt + c) * sizeof(_Float16));
      async_g2l_b128(loff, goff, ctx);
    }
    // B: 64x32 fp32 -> f16 (needs convert, stays on VGPR path).
#pragma unroll
    for (int j = 0; j < 2; ++j) {
      const int f = tid + j * 256;
      const int r = f >> 3, c = (f & 7) << 2;
      const float4 d = *(const float4*)(w_out + (size_t)(n0 + r) * A_DIM + kt + c);
      _Float16* p = &lB[r * LDA + c];
      p[0] = (_Float16)d.x; p[1] = (_Float16)d.y;
      p[2] = (_Float16)d.z; p[3] = (_Float16)d.w;
    }
    wait_asynccnt0();
    __syncthreads();
    v16h af[2], bf[2];
#pragma unroll
    for (int mi = 0; mi < 2; ++mi)
      af[mi] = frag_a(&lA[(wmi * 32 + mi * 16) * LDA], LDA, lane);
#pragma unroll
    for (int ni = 0; ni < 2; ++ni)
      bf[ni] = frag_b(&lB[(wni * 32 + ni * 16) * LDA], LDA, lane);
#pragma unroll
    for (int mi = 0; mi < 2; ++mi)
#pragma unroll
      for (int ni = 0; ni < 2; ++ni)
        acc[mi][ni] = wmma16(af[mi], bf[ni], acc[mi][ni]);
  }
  const int col = lane & 15, hs = lane >> 4;
#pragma unroll
  for (int mi = 0; mi < 2; ++mi)
#pragma unroll
    for (int ni = 0; ni < 2; ++ni)
#pragma unroll
      for (int i = 0; i < 8; ++i)
        out[(size_t)(gm0 + wmi * 32 + mi * 16 + i + 8 * hs) * D_DIM +
            n0 + wni * 32 + ni * 16 + col] = acc[mi][ni][i];
}

extern "C" void kernel_launch(void* const* d_in, const int* in_sizes, int n_in,
                              void* d_out, int out_size, void* d_ws, size_t ws_size,
                              hipStream_t stream) {
  const float* x     = (const float*)d_in[0];  // [2048,1024]
  const float* w_in  = (const float*)d_in[1];  // [3072,1024]
  const float* w_out = (const float*)d_in[2];  // [1024,1024]
  (void)in_sizes; (void)n_in; (void)out_size; (void)ws_size;
  float* out = (float*)d_out;                  // [2048,1024] f32

  // Workspace: Q,K,V in [H][S][hd] f16 (4 MiB each) + ctx [S][A] f16 (4 MiB).
  const size_t hsz = (size_t)H_NUM * S_LEN * HD;
  _Float16* qh  = (_Float16*)d_ws;
  _Float16* kh  = qh + hsz;
  _Float16* vh  = kh + hsz;
  _Float16* ctx = vh + hsz;

  qkv_gemm_k<<<dim3(S_LEN / 128, 48), 256, 0, stream>>>(x, w_in, qh, kh, vh);
  attn_k<<<dim3(S_LEN / 64, H_NUM), 128, 0, stream>>>(qh, kh, vh, ctx);
  out_gemm_k<<<dim3(S_LEN / 128, D_DIM / 64), 256, 0, stream>>>(ctx, w_out, out);
}
